// KnnExact_18554258719443
// MI455X (gfx1250) — matrix operand order repaired and checked by reading, not compiled, not run on hardware
//
#include <hip/hip_runtime.h>
#include <hip/hip_bf16.h>

// ---------------------------------------------------------------------------
// Exact self-kNN (N=16384, D=256, k=32) for gfx1250 / MI455X.
//  * Gram matrix via split-bf16 WMMA (hi*hi + hi*lo + lo*hi, f32 accumulate)
//  * waves partition ROWS; all 8 waves share each 32-column B chunk -> B is
//    served from WGP$, unique L2 traffic = N^2*D*4B / 128 rows (~2.1 GB)
//  * A tile pre-packed into LDS in WMMA fragment layout (128 KB/block),
//    loaded from global exactly once
//  * 6 independent WMMA accumulators (2 col-subtiles x 3 split terms)
//  * top-32 slots in wave-private LDS (touched only on rare inserts, ~200 per
//    row total); only the 16 hot per-row thresholds stay in VGPRs -> kernel
//    stays under 256 VGPRs, no s_set_vgpr_msb churn
//  * zero block barriers in the main loop
//  * exact fp32 distance recompute pass at the end (matches reference path)
// ---------------------------------------------------------------------------

typedef __attribute__((ext_vector_type(16))) __bf16 v16bf;
typedef __attribute__((ext_vector_type(8)))  __bf16 v8bf;
typedef __attribute__((ext_vector_type(8)))  float  v8f;

constexpr int NN = 16384;
constexpr int DD = 256;
constexpr int KK = 32;
constexpr int RW = 16;       // rows per wave (one WMMA M tile)
constexpr int WV = 8;        // waves per block
constexpr int RB = RW * WV;  // 128 rows per block
constexpr int CT = 32;       // columns per chunk (2 WMMA N tiles)

// ---------------------------------------------------------------------------
// Kernel 1: split fp32 rows into bf16 hi/lo planes + squared norms.
// ---------------------------------------------------------------------------
__global__ __launch_bounds__(256) void knn_prep(const float* __restrict__ x,
                                                __bf16* __restrict__ xhi,
                                                __bf16* __restrict__ xlo,
                                                float* __restrict__ sq) {
  __shared__ float red[256];
  const int r = blockIdx.x;
  const int e = threadIdx.x;
  const float f = x[(size_t)r * DD + e];
  const __bf16 h = (__bf16)f;
  xhi[(size_t)r * DD + e] = h;
  xlo[(size_t)r * DD + e] = (__bf16)(f - (float)h);
  red[e] = f * f;
  __syncthreads();
  #pragma unroll
  for (int s = 128; s > 0; s >>= 1) {
    if (e < s) red[e] += red[e + s];
    __syncthreads();
  }
  if (e == 0) sq[r] = red[0];
}

// ---------------------------------------------------------------------------
// Wave32 reductions.
// ---------------------------------------------------------------------------
__device__ __forceinline__ float wave_max(float v) {
  #pragma unroll
  for (int off = 16; off >= 1; off >>= 1) v = fmaxf(v, __shfl_xor(v, off, 32));
  return v;
}

__device__ __forceinline__ int wave_argmax(float v, int lane) {
  float bv = v;
  int   bl = lane;
  #pragma unroll
  for (int off = 16; off >= 1; off >>= 1) {
    float ov = __shfl_xor(bv, off, 32);
    int   ol = __shfl_xor(bl, off, 32);
    if (ov > bv || (ov == bv && ol < bl)) { bv = ov; bl = ol; }
  }
  return bl;
}

// ---------------------------------------------------------------------------
// Kernel 2: fused GEMM + running top-k.  256 threads = 8 waves; wave w owns
// rows [blockIdx*128 + w*16, +16).  No block-level barriers anywhere.
// ---------------------------------------------------------------------------
__global__ __launch_bounds__(256) void knn_main(const __bf16* __restrict__ xhi,
                                                const __bf16* __restrict__ xlo,
                                                const float* __restrict__ sq,
                                                float* __restrict__ out) {
  // A fragments, pre-packed per-lane: [wave][kt][hi/lo][lane][16 bf16] = 128KB
  __shared__ __align__(32) __bf16 ldsA[WV][8][2][32][16];
  __shared__ float          ldsDist[WV][RW][CT + 1];  // +1 pad, 17KB
  __shared__ float          ldsBd[WV][RW][32];        // slot distances, 16KB
  __shared__ unsigned short ldsBi[WV][RW][32];        // slot indices,   8KB

  const int tid   = threadIdx.x;
  const int lane  = tid & 31;
  const int w     = tid >> 5;                  // wave 0..7
  const int m0    = blockIdx.x * RB + w * RW;  // this wave's row base
  const int nloc  = lane & 15;                 // M (A) / N (B) within tile
  const int khalf = lane >> 4;                 // K-half this lane feeds
  const int mofs  = (lane >= 16) ? 8 : 0;      // C/D layout M offset

  // ---- A tile: global -> LDS fragments, exactly once ----------------------
  {
    const __bf16* aHiBase = xhi + (size_t)(m0 + nloc) * DD;
    const __bf16* aLoBase = xlo + (size_t)(m0 + nloc) * DD;
    #pragma unroll
    for (int kt = 0; kt < 8; ++kt) {
      const int k0 = kt * 32 + khalf * 16;
      v8bf a0h = *(const v8bf*)(aHiBase + k0);
      v8bf a1h = *(const v8bf*)(aHiBase + k0 + 8);
      v8bf a0l = *(const v8bf*)(aLoBase + k0);
      v8bf a1l = *(const v8bf*)(aLoBase + k0 + 8);
      v16bf ah, al;
      #pragma unroll
      for (int i = 0; i < 8; ++i) {
        ah[i] = a0h[i]; ah[8 + i] = a1h[i];
        al[i] = a0l[i]; al[8 + i] = a1l[i];
      }
      *(v16bf*)&ldsA[w][kt][0][lane][0] = ah;
      *(v16bf*)&ldsA[w][kt][1][lane][0] = al;
    }
  }

  float sqr[8];
  #pragma unroll
  for (int v = 0; v < 8; ++v) sqr[v] = sq[m0 + v + mofs];

  // GEMM for one 32-column chunk; distances land in this wave's LDS panel.
  auto compute_chunk = [&](int c0) {
    const __bf16* b0h = xhi + (size_t)(c0 + nloc) * DD + khalf * 16;
    const __bf16* b0l = xlo + (size_t)(c0 + nloc) * DD + khalf * 16;
    const __bf16* b1h = b0h + (size_t)16 * DD;
    const __bf16* b1l = b0l + (size_t)16 * DD;

    v8f a0hh = {}, a0hl = {}, a0lh = {};
    v8f a1hh = {}, a1hl = {}, a1lh = {};
    #pragma unroll
    for (int kt = 0; kt < 8; ++kt) {
      const int k0 = kt * 32;
      v16bf Ah = *(const v16bf*)&ldsA[w][kt][0][lane][0];
      v16bf Al = *(const v16bf*)&ldsA[w][kt][1][lane][0];
      v16bf B0h = *(const v16bf*)(b0h + k0);
      v16bf B0l = *(const v16bf*)(b0l + k0);
      v16bf B1h = *(const v16bf*)(b1h + k0);
      v16bf B1l = *(const v16bf*)(b1l + k0);
      // 6 independent accumulator chains -> dep distance 6 between WMMAs.
      a0hh = __builtin_amdgcn_wmma_f32_16x16x32_bf16(false, Ah, false, B0h, (short)0, a0hh, false, false);
      a1hh = __builtin_amdgcn_wmma_f32_16x16x32_bf16(false, Ah, false, B1h, (short)0, a1hh, false, false);
      a0hl = __builtin_amdgcn_wmma_f32_16x16x32_bf16(false, Ah, false, B0l, (short)0, a0hl, false, false);
      a1hl = __builtin_amdgcn_wmma_f32_16x16x32_bf16(false, Ah, false, B1l, (short)0, a1hl, false, false);
      a0lh = __builtin_amdgcn_wmma_f32_16x16x32_bf16(false, Al, false, B0h, (short)0, a0lh, false, false);
      a1lh = __builtin_amdgcn_wmma_f32_16x16x32_bf16(false, Al, false, B1h, (short)0, a1lh, false, false);
    }

    const float sqc0 = sq[c0 + nloc];
    const float sqc1 = sq[c0 + 16 + nloc];
    #pragma unroll
    for (int v = 0; v < 8; ++v) {
      const float d0 = sqr[v] + sqc0 - 2.0f * (a0hh[v] + a0hl[v] + a0lh[v]);
      const float d1 = sqr[v] + sqc1 - 2.0f * (a1hh[v] + a1hl[v] + a1lh[v]);
      ldsDist[w][v + mofs][nloc]      = d0;
      ldsDist[w][v + mofs][16 + nloc] = d1;
    }
  };

  // ---- seed chunk: first 32 candidates fill the 32 slots directly ---------
  float cur_max[RW];
  compute_chunk(0);
  #pragma unroll
  for (int r = 0; r < RW; ++r) {
    const float d = ldsDist[w][r][lane];
    ldsBd[w][r][lane] = d;
    ldsBi[w][r][lane] = (unsigned short)lane;
    cur_max[r] = wave_max(d);
  }

  // ---- steady state --------------------------------------------------------
  for (int c0 = CT; c0 < NN; c0 += CT) {
    compute_chunk(c0);
    const int gi = c0 + lane;
    #pragma unroll
    for (int r = 0; r < RW; ++r) {
      const float d = ldsDist[w][r][lane];
      unsigned mask = (unsigned)__builtin_amdgcn_ballot_w32(d < cur_max[r]);
      while (mask) {  // rare: ~k*ln(N/k) insertions per row over whole run
        const int src = __builtin_ctz(mask);
        mask &= mask - 1;
        const float dj = __shfl(d, src, 32);
        const int   ij = __shfl(gi, src, 32);
        if (dj < cur_max[r]) {
          float slot = ldsBd[w][r][lane];
          const int ml = wave_argmax(slot, lane);
          if (lane == ml) {
            ldsBd[w][r][lane] = dj;
            ldsBi[w][r][lane] = (unsigned short)ij;
            slot = dj;
          }
          cur_max[r] = wave_max(slot);
        }
      }
    }
  }

  // ---- per row: bitonic sort ascending across lanes, then write -----------
  #pragma unroll 1
  for (int r = 0; r < RW; ++r) {
    float bd = ldsBd[w][r][lane];
    int   bi = (int)ldsBi[w][r][lane];
    #pragma unroll
    for (int size = 2; size <= 32; size <<= 1) {
      #pragma unroll
      for (int stride = size >> 1; stride > 0; stride >>= 1) {
        const float od = __shfl_xor(bd, stride, 32);
        const int   oi = __shfl_xor(bi, stride, 32);
        const bool take_min = (((lane & stride) == 0) == ((lane & size) == 0));
        const bool other_sm = (od < bd) || (od == bd && oi < bi);
        if (other_sm == take_min) { bd = od; bi = oi; }
      }
    }
    const int row = m0 + r;
    out[(size_t)row * KK + lane]                   = (float)bi;  // indices
    out[(size_t)NN * KK + (size_t)row * KK + lane] = bd;         // distances
  }
}

// ---------------------------------------------------------------------------
// Kernel 3: exact fp32 recompute of selected distances (reference's
// sum((xi - x[idx])**2) path). One wave per (row, slot).
// ---------------------------------------------------------------------------
__global__ __launch_bounds__(256) void knn_refine(const float* __restrict__ x,
                                                  float* __restrict__ out) {
  const int lane = threadIdx.x & 31;
  const int w    = threadIdx.x >> 5;
  const long g   = (long)blockIdx.x * 8 + w;       // 0 .. N*K-1
  const int row  = (int)(g / KK);
  int j = (int)out[g];
  j = j < 0 ? 0 : (j >= NN ? NN - 1 : j);
  const float* xr = x + (size_t)row * DD;
  const float* xj = x + (size_t)j * DD;
  float s = 0.0f;
  #pragma unroll
  for (int e = lane; e < DD; e += 32) {
    const float diff = xr[e] - xj[e];
    s += diff * diff;
  }
  #pragma unroll
  for (int off = 16; off >= 1; off >>= 1) s += __shfl_xor(s, off, 32);
  if (lane == 0) out[(size_t)NN * KK + g] = s;
}

// ---------------------------------------------------------------------------
extern "C" void kernel_launch(void* const* d_in, const int* in_sizes, int n_in,
                              void* d_out, int out_size, void* d_ws, size_t ws_size,
                              hipStream_t stream) {
  const float* x = (const float*)d_in[0];
  // d_in[1] is k (== 32), baked in at compile time.
  (void)in_sizes; (void)n_in; (void)out_size; (void)ws_size;

  char* ws = (char*)d_ws;
  __bf16* xhi = (__bf16*)ws;                                   // N*D*2 bytes
  __bf16* xlo = (__bf16*)(ws + (size_t)NN * DD * 2);           // N*D*2 bytes
  float*  sq  = (float*)(ws + (size_t)NN * DD * 4);            // N*4   bytes
  float*  out = (float*)d_out;

  knn_prep<<<NN, 256, 0, stream>>>(x, xhi, xlo, sq);
  knn_main<<<NN / RB, 256, 0, stream>>>(xhi, xlo, sq, out);
  knn_refine<<<(NN * KK) / 8, 256, 0, stream>>>(x, out);
}
